// HybridGNNLSTMClassifier_8933531975923
// MI455X (gfx1250) — matrix operand order repaired and checked by reading, not compiled, not run on hardware
//
#include <hip/hip_runtime.h>
#include <hip/hip_bf16.h>
#include <math.h>

typedef __attribute__((ext_vector_type(16))) _Float16 v16h;
typedef __attribute__((ext_vector_type(8)))  float    v8f;

#define NN      131072      // B_GRAPHS * T_NODES
#define EDGES   2097152
#define TSTEPS  2048
#define BGR     64
#define GH      32          // GNN hidden
#define LH      16          // LSTM hidden

__device__ __forceinline__ float sigm(float x) { return 1.0f / (1.0f + expf(-x)); }

// ---------------------------------------------------------------- init
__global__ void k_init(float* deg, float* S1, float* acc2) {
    int idx = blockIdx.x * 256 + threadIdx.x;
    if (idx < NN) { deg[idx] = 1.0f; S1[idx] = 0.0f; }   // deg starts at 1 (self-loop)
    if (idx < NN * GH) acc2[idx] = 0.0f;
}

// ---------------------------------------------------------------- degree scatter
__global__ void k_deg(const int* __restrict__ dst, float* deg) {
    int e = blockIdx.x * 256 + threadIdx.x;
    if (e < EDGES) atomicAdd(&deg[dst[e]], 1.0f);
}

// ---------------------------------------------------------------- dinv + pre-scaled x
__global__ void k_dinv(const float* __restrict__ x, const float* __restrict__ deg,
                       float* dinv, float* xs0) {
    int i = blockIdx.x * 256 + threadIdx.x;
    if (i < NN) {
        float d = deg[i];
        float r = (d > 0.0f) ? rsqrtf(d) : 0.0f;
        dinv[i] = r;
        xs0[i]  = x[i] * r;        // IN_C == 1: scalar per node
    }
}

// ---------------------------------------------------------------- GCN1 scatter (scalar per edge!)
__global__ void k_scatter1(const int* __restrict__ src, const int* __restrict__ dst,
                           const float* __restrict__ xs0, float* S1) {
    int e = blockIdx.x * 256 + threadIdx.x;
    if (e < EDGES) atomicAdd(&S1[dst[e]], xs0[src[e]]);
}

// ---------------------------------------------------------------- h1 = relu((S1+self)*dinv*W1 + b1)
__global__ void k_h1(const float* __restrict__ S1, const float* __restrict__ xs0,
                     const float* __restrict__ dinv, const float* __restrict__ W1,
                     const float* __restrict__ b1, float* __restrict__ h1) {
    int i = blockIdx.x * 256 + threadIdx.x;
    if (i >= NN) return;
    float a = (S1[i] + xs0[i]) * dinv[i];   // scalar aggregate for node i
    #pragma unroll
    for (int c = 0; c < GH; ++c) {
        float v = fmaf(a, W1[c], b1[c]);
        h1[i * GH + c] = v > 0.0f ? v : 0.0f;
    }
}

// ---------------------------------------------------------------- GEMM: xs2 = (h1 @ W2) * dinv  via WMMA
// Each wave: one 16-row tile, two 16x16 col tiles, K=32 in one v_wmma_f32_16x16x32_f16 each.
__global__ void k_gemm2(const float* __restrict__ h1, const float* __restrict__ W2,
                        const float* __restrict__ dinv, float* __restrict__ xs2) {
    __shared__ _Float16 w2s[GH * GH];
    int tid = threadIdx.x;
    for (int p = tid; p < GH * GH; p += 256) w2s[p] = (_Float16)W2[p];
    __syncthreads();

    int wave = tid >> 5;
    int lane = tid & 31;
    int r    = lane & 15;          // A: row-in-tile, B/C: col-in-tile
    int hi   = lane >> 4;          // lane half-group
    int rowBase = (blockIdx.x * 8 + wave) * 16;

    // A tile (16x32 f16): lanes 0-15 hold k 0..15; lanes 16-31 hold k 16..31
    const float* arow = h1 + (rowBase + r) * GH;
    int klo = hi ? 16 : 0;
    v16h a;
    #pragma unroll
    for (int j = 0; j < 8; ++j) a[j]     = (_Float16)arow[klo + j];
    #pragma unroll
    for (int j = 0; j < 8; ++j) a[8 + j] = (_Float16)arow[klo + 8 + j];

    int kb = hi ? 16 : 0;          // B: lanes 0-15 k=0..15, lanes 16-31 k=16..31
    #pragma unroll
    for (int t = 0; t < 2; ++t) {
        v16h b;
        #pragma unroll
        for (int j = 0; j < 16; ++j) b[j] = w2s[(kb + j) * GH + t * 16 + r];
        v8f c = {};
        c = __builtin_amdgcn_wmma_f32_16x16x32_f16(false, a, false, b, (short)0, c, false, false);
        #pragma unroll
        for (int q = 0; q < 8; ++q) {
            int row = rowBase + q + (hi ? 8 : 0);
            xs2[row * GH + t * 16 + r] = c[q] * dinv[row];   // pre-scale by dinv[src]
        }
    }
}

// ---------------------------------------------------------------- GCN2 scatter (32 channels/edge)
__global__ void k_scatter2(const int* __restrict__ src, const int* __restrict__ dst,
                           const float* __restrict__ xs2, float* acc2) {
    int idx = blockIdx.x * 256 + threadIdx.x;
    int e = idx >> 5;
    int c = idx & 31;
    if (e < EDGES) {
        int s = src[e], d = dst[e];
        atomicAdd(&acc2[d * GH + c], xs2[s * GH + c]);
    }
}

// ---------------------------------------------------------------- h2 = (acc2 + self)*dinv + b2
__global__ void k_h2(const float* __restrict__ acc2, const float* __restrict__ xs2,
                     const float* __restrict__ dinv, const float* __restrict__ b2,
                     float* __restrict__ h2) {
    int idx = blockIdx.x * 256 + threadIdx.x;
    if (idx >= NN * GH) return;
    int i = idx >> 5, c = idx & 31;
    h2[idx] = (acc2[idx] + xs2[idx]) * dinv[i] + b2[c];
}

// ---------------------------------------------------------------- LSTM: 4 blocks x 16 seqs, 4 waves = 4 gate tiles
// Per step: gates[16,64] = X[16,32]@Wih^T + Hpad[16,32]@Whh^T  (2 WMMAs/wave, weights in regs)
__global__ void k_lstm(const float* __restrict__ h2,
                       const float* __restrict__ W_ih, const float* __restrict__ W_hh,
                       const float* __restrict__ b_ih, const float* __restrict__ b_hh,
                       float* __restrict__ hlast) {
    __shared__ _Float16 Xs[16 * GH];      // current step inputs, f16
    __shared__ _Float16 Hs[16 * LH];      // hidden state, f16
    __shared__ float    Gs[4 * 16 * LH];  // gate tiles [gate][seq][unit]

    int tid  = threadIdx.x;               // 0..127
    int wave = tid >> 5;                  // gate index 0..3 (i,f,g,o)
    int lane = tid & 31;
    int r    = lane & 15;
    int hi   = lane >> 4;
    int seq0 = blockIdx.x * 16;

    // --- load weight tiles into registers once (B layout: lane=col, VGPR-pair=k)
    int colg = wave * LH + r;             // gate output column 0..63
    int kb   = hi ? 8 : 0;
    v16h bx, bh;
    #pragma unroll
    for (int j = 0; j < 16; ++j) bx[j] = (_Float16)W_ih[colg * GH + kb + j + (j >= 8 ? 8 : 0)];
    #pragma unroll
    for (int j = 0; j < 16; ++j) {
        int k = kb + j + (j >= 8 ? 8 : 0);
        bh[j] = (k < LH) ? (_Float16)W_hh[colg * LH + k] : (_Float16)0.0f;
    }

    // --- per-thread states: s0 = tid (seqs 0..7), s1 = tid+128 (seqs 8..15)
    int sj = tid & 15;
    int sb0 = tid >> 4, sb1 = 8 + (tid >> 4);
    float bi0[4], bi1[4];
    #pragma unroll
    for (int g = 0; g < 4; ++g) {
        bi0[g] = b_ih[g * LH + sj] + b_hh[g * LH + sj];
        bi1[g] = bi0[g];                  // same unit j, same bias
    }
    float c0 = 0.f, c1 = 0.f, hh0 = 0.f, hh1 = 0.f;
    Hs[tid] = (_Float16)0.0f; Hs[tid + 128] = (_Float16)0.0f;
    __syncthreads();

    for (int t = 0; t < TSTEPS; ++t) {
        // stage X_t into LDS as f16 (512 elems / 128 threads); prefetch next row (1 cacheline)
        #pragma unroll
        for (int p = 0; p < 4; ++p) {
            int idx = tid + p * 128;
            int row = idx >> 5, col = idx & 31;
            const float* gp = h2 + ((size_t)(seq0 + row) * TSTEPS + t) * GH + col;
            Xs[idx] = (_Float16)(*gp);
            if (col == 0 && t + 1 < TSTEPS)
                __builtin_prefetch(gp + GH, 0, 0);   // global_prefetch_b8, next timestep row
        }
        __syncthreads();   // Xs ready; prev Gs reads + Hs writes complete

        // A tiles from LDS
        int klo = hi ? 8 : 0;
        v16h ax, ah;
        #pragma unroll
        for (int j = 0; j < 8; ++j) ax[j]     = Xs[r * GH + klo + j];
        #pragma unroll
        for (int j = 0; j < 8; ++j) ax[8 + j] = Xs[r * GH + 16 + klo + j];
        #pragma unroll
        for (int j = 0; j < 8; ++j) ah[j]     = Hs[r * LH + klo + j];  // k = klo+j < 16
        #pragma unroll
        for (int j = 0; j < 8; ++j) ah[8 + j] = (_Float16)0.0f;        // padded k >= 16

        v8f acc = {};
        acc = __builtin_amdgcn_wmma_f32_16x16x32_f16(false, ax, false, bx, (short)0, acc, false, false);
        acc = __builtin_amdgcn_wmma_f32_16x16x32_f16(false, ah, false, bh, (short)0, acc, false, false);

        #pragma unroll
        for (int q = 0; q < 8; ++q) {
            int b = q + (hi ? 8 : 0);
            Gs[wave * 256 + b * LH + r] = acc[q];
        }
        __syncthreads();   // gates ready

        // c/h update: i,f,g,o gathered across gate tiles
        {
            float ig = Gs[0 * 256 + tid] + bi0[0];
            float fg = Gs[1 * 256 + tid] + bi0[1];
            float gg = Gs[2 * 256 + tid] + bi0[2];
            float og = Gs[3 * 256 + tid] + bi0[3];
            c0  = sigm(fg) * c0 + sigm(ig) * tanhf(gg);
            hh0 = sigm(og) * tanhf(c0);
            Hs[tid] = (_Float16)hh0;

            int s1 = tid + 128;
            ig = Gs[0 * 256 + s1] + bi1[0];
            fg = Gs[1 * 256 + s1] + bi1[1];
            gg = Gs[2 * 256 + s1] + bi1[2];
            og = Gs[3 * 256 + s1] + bi1[3];
            c1  = sigm(fg) * c1 + sigm(ig) * tanhf(gg);
            hh1 = sigm(og) * tanhf(c1);
            Hs[s1] = (_Float16)hh1;
        }
        // next iteration's __syncthreads (after X staging) orders Gs-overwrite vs reads
    }
    hlast[(seq0 + sb0) * LH + sj] = hh0;
    hlast[(seq0 + sb1) * LH + sj] = hh1;
}

// ---------------------------------------------------------------- final FC + sigmoid
__global__ void k_fc(const float* __restrict__ hlast, const float* __restrict__ Wfc,
                     const float* __restrict__ bfc, float* __restrict__ out) {
    int b = threadIdx.x;
    if (b < BGR) {
        float v = bfc[0];
        #pragma unroll
        for (int j = 0; j < LH; ++j) v = fmaf(hlast[b * LH + j], Wfc[j], v);
        out[b] = sigm(v);
    }
}

// ================================================================ launch
extern "C" void kernel_launch(void* const* d_in, const int* in_sizes, int n_in,
                              void* d_out, int out_size, void* d_ws, size_t ws_size,
                              hipStream_t stream) {
    const float* x    = (const float*)d_in[0];
    const int*   ei   = (const int*)d_in[1];      // [2, E]
    const int*   src  = ei;
    const int*   dst  = ei + EDGES;
    const float* W1   = (const float*)d_in[3];
    const float* b1   = (const float*)d_in[4];
    const float* W2   = (const float*)d_in[5];
    const float* b2   = (const float*)d_in[6];
    const float* W_ih = (const float*)d_in[7];
    const float* W_hh = (const float*)d_in[8];
    const float* b_ih = (const float*)d_in[9];
    const float* b_hh = (const float*)d_in[10];
    const float* Wfc  = (const float*)d_in[11];
    const float* bfc  = (const float*)d_in[12];
    float* out = (float*)d_out;

    // workspace layout (floats)
    float* ws    = (float*)d_ws;
    float* deg   = ws;                    // N
    float* dinv  = deg  + NN;             // N
    float* xs0   = dinv + NN;             // N
    float* S1    = xs0  + NN;             // N
    float* h1    = S1   + NN;             // N*32  (reused as h2 after GEMM)
    float* xs2   = h1   + (size_t)NN * GH;// N*32
    float* acc2  = xs2  + (size_t)NN * GH;// N*32
    float* hlast = acc2 + (size_t)NN * GH;// 64*16
    float* h2    = h1;                    // alias: h1 dead after k_gemm2

    int thr = 256;
    k_init    <<<(NN * GH + thr - 1) / thr, thr, 0, stream>>>(deg, S1, acc2);
    k_deg     <<<(EDGES + thr - 1) / thr, thr, 0, stream>>>(dst, deg);
    k_dinv    <<<(NN + thr - 1) / thr, thr, 0, stream>>>(x, deg, dinv, xs0);
    k_scatter1<<<(EDGES + thr - 1) / thr, thr, 0, stream>>>(src, dst, xs0, S1);
    k_h1      <<<(NN + thr - 1) / thr, thr, 0, stream>>>(S1, xs0, dinv, W1, b1, h1);
    k_gemm2   <<<NN / (16 * 8), thr, 0, stream>>>(h1, W2, dinv, xs2);
    k_scatter2<<<((size_t)EDGES * GH + thr - 1) / thr, thr, 0, stream>>>(src, dst, xs2, acc2);
    k_h2      <<<(NN * GH + thr - 1) / thr, thr, 0, stream>>>(acc2, xs2, dinv, b2, h2);
    k_lstm    <<<BGR / 16, 128, 0, stream>>>(h2, W_ih, W_hh, b_ih, b_hh, hlast);
    k_fc      <<<1, 64, 0, stream>>>(hlast, Wfc, bfc, out);
}
